// QuantLinear_38053410242621
// MI455X (gfx1250) — 8-bit matrix operand order repaired and checked by reading, compiled, not run on hardware
//
#include <hip/hip_runtime.h>
#include <hip/hip_bf16.h>
#include <stdint.h>

typedef __attribute__((ext_vector_type(16))) int   v16i;
typedef __attribute__((ext_vector_type(8)))  int   v8i;
typedef __attribute__((ext_vector_type(4)))  int   v4i;
typedef __attribute__((ext_vector_type(8)))  float v8f;
typedef __attribute__((ext_vector_type(4)))  float v4f;

#define M_DIM 8192
#define N_DIM 4096
#define K_DIM 4096
#define NKB   (K_DIM / 128)

#if __has_builtin(__builtin_amdgcn_global_load_async_to_lds_b128)
#define ASYNC_OK 1
#else
#define ASYNC_OK 0
#endif

#define AS1G(p) ((v4i*)(p))
#define AS3L(p) ((v4i*)(p))

__device__ inline unsigned enc_e4m3(float x, float* dec) {
    unsigned u = __float_as_uint(x);
    unsigned s = (u >> 24) & 0x80u;
    float ax = fabsf(x);
    if (ax >= 448.0f) { *dec = (x < 0.f) ? -448.f : 448.f; return s | 0x7Eu; }
    if (ax < 0.015625f) {
        unsigned m = (unsigned)rintf(ax * 512.0f);
        float d = (float)m * 0.001953125f;
        *dec = (x < 0.f) ? -d : d;
        return (m >= 8u) ? (s | 0x08u) : (s | m);
    }
    unsigned ur = (u + 0x0007FFFFu + ((u >> 20) & 1u)) & 0xFFF00000u;
    *dec = __uint_as_float(ur);
    unsigned e  = (ur >> 23) & 0xFFu;
    unsigned m3 = (ur >> 20) & 7u;
    return s | ((e - 120u) << 3) | m3;
}

__global__ __launch_bounds__(256) void qk_abs_partial(const float* __restrict__ W,
                                                      float* __restrict__ partials, int n) {
    __shared__ float red[256];
    float s = 0.f;
    const int stride = gridDim.x * blockDim.x;
    for (int i = blockIdx.x * blockDim.x + threadIdx.x; i < n; i += stride)
        s += fabsf(W[i]);
    red[threadIdx.x] = s;
    __syncthreads();
    for (int off = 128; off > 0; off >>= 1) {
        if ((int)threadIdx.x < off) red[threadIdx.x] += red[threadIdx.x + off];
        __syncthreads();
    }
    if (threadIdx.x == 0) partials[blockIdx.x] = red[0];
}

__global__ __launch_bounds__(256) void qk_abs_final(const float* __restrict__ partials,
                                                    float* __restrict__ scale_out,
                                                    int nparts, float inv_count) {
    __shared__ float red[256];
    float s = 0.f;
    for (int i = threadIdx.x; i < nparts; i += 256) s += partials[i];
    red[threadIdx.x] = s;
    __syncthreads();
    for (int off = 128; off > 0; off >>= 1) {
        if ((int)threadIdx.x < off) red[threadIdx.x] += red[threadIdx.x + off];
        __syncthreads();
    }
    if (threadIdx.x == 0) scale_out[0] = fmaxf(red[0] * inv_count, 1e-8f);
}

__global__ __launch_bounds__(256) void qk_quant_w8(const float* __restrict__ W,
                                                   const float* __restrict__ scale_p,
                                                   unsigned char* __restrict__ W8, int n4) {
    const float inv = 1.0f / scale_p[0];
    const int stride = gridDim.x * blockDim.x;
    for (int i = blockIdx.x * blockDim.x + threadIdx.x; i < n4; i += stride) {
        v4f w = ((const v4f*)W)[i];
        float d;
        unsigned b0 = enc_e4m3(fminf(fmaxf(rintf(w[0] * inv), -2.f), 2.f), &d);
        unsigned b1 = enc_e4m3(fminf(fmaxf(rintf(w[1] * inv), -2.f), 2.f), &d);
        unsigned b2 = enc_e4m3(fminf(fmaxf(rintf(w[2] * inv), -2.f), 2.f), &d);
        unsigned b3 = enc_e4m3(fminf(fmaxf(rintf(w[3] * inv), -2.f), 2.f), &d);
        ((unsigned*)W8)[i] = b0 | (b1 << 8) | (b2 << 16) | (b3 << 24);
    }
}

__global__ __launch_bounds__(256) void qk_split_x8(const float* __restrict__ X,
                                                   unsigned char* __restrict__ XHI,
                                                   unsigned char* __restrict__ XLO) {
    const unsigned tid  = blockIdx.x * 256 + threadIdx.x;
    const unsigned t    = tid >> 8;
    const unsigned idx  = tid & 255u;
    const unsigned lane = idx >> 3;
    const unsigned c    = idx & 7u;
    const unsigned mt   = t >> 5, kb = t & 31u;
    const unsigned m    = mt * 16 + (lane & 15u);
    const unsigned k    = kb * 128 + 32u * (c >> 1) + 8u * (c & 1u) + 16u * (lane >> 4);

    const float* src = X + (size_t)m * K_DIM + k;
    v4f x0 = *(const v4f*)(src);
    v4f x1 = *(const v4f*)(src + 4);

    unsigned hi = 0, hi2 = 0, lo = 0, lo2 = 0;
#pragma unroll
    for (int e = 0; e < 4; ++e) {
        float d;
        unsigned hb = enc_e4m3(x0[e], &d);
        unsigned lb = enc_e4m3((x0[e] - d) * 16.0f, &d);
        hi |= hb << (8 * e);  lo |= lb << (8 * e);
    }
#pragma unroll
    for (int e = 0; e < 4; ++e) {
        float d;
        unsigned hb = enc_e4m3(x1[e], &d);
        unsigned lb = enc_e4m3((x1[e] - d) * 16.0f, &d);
        hi2 |= hb << (8 * e); lo2 |= lb << (8 * e);
    }
    ((unsigned*)XHI)[2 * tid]     = hi;
    ((unsigned*)XHI)[2 * tid + 1] = hi2;
    ((unsigned*)XLO)[2 * tid]     = lo;
    ((unsigned*)XLO)[2 * tid + 1] = lo2;
}

#define A_LSTR 80
#define A_TSTR (32 * A_LSTR)
#define B_STR  144
#define AH_OFF 0
#define AL_OFF (8 * A_TSTR)
#define B_OFF  (16 * A_TSTR)
#define BUF_BYTES (16 * A_TSTR + 128 * B_STR)

extern __shared__ __align__(128) unsigned char qk_smem[];

__device__ __forceinline__ void qk_stage(const unsigned char* gh, const unsigned char* gl,
                                         const unsigned char* gw,
                                         unsigned char* dh, unsigned char* dl,
                                         unsigned char* db) {
#pragma unroll
    for (int q = 0; q < 4; ++q) {
#if ASYNC_OK
        __builtin_amdgcn_global_load_async_to_lds_b128(AS1G(gh + q * 16), AS3L(dh + q * 16), 0, 0);
        __builtin_amdgcn_global_load_async_to_lds_b128(AS1G(gl + q * 16), AS3L(dl + q * 16), 0, 0);
        __builtin_amdgcn_global_load_async_to_lds_b128(AS1G(gw + q * 16), AS3L(db + q * 16), 0, 0);
#else
        *(v4i*)(dh + q * 16) = *(const v4i*)(gh + q * 16);
        *(v4i*)(dl + q * 16) = *(const v4i*)(gl + q * 16);
        *(v4i*)(db + q * 16) = *(const v4i*)(gw + q * 16);
#endif
    }
}

__global__ __launch_bounds__(256) void qk_gemm_fp8(const unsigned char* __restrict__ XHI,
                                                   const unsigned char* __restrict__ XLO,
                                                   const unsigned char* __restrict__ W8,
                                                   const float* __restrict__ scale_p,
                                                   float* __restrict__ Out) {
    const int tid  = threadIdx.x;
    const int lane = tid & 31;
    const int wave = tid >> 5;
    const int wr   = wave >> 2;
    const int wc   = wave & 3;
    const int l16  = lane & 15;
    const int lh   = lane >> 4;

    const int n0  = blockIdx.x * 128;
    const int mt0 = blockIdx.y * 8;

    const int ss = tid >> 5;
    const int sl = tid & 31;
    const int bn = tid >> 1;
    const int bh = (tid & 1) * 64;

    const unsigned char* ghb = XHI + ((size_t)(mt0 + ss) * NKB) * 2048 + (size_t)sl * 64;
    const unsigned char* glb = XLO + ((size_t)(mt0 + ss) * NKB) * 2048 + (size_t)sl * 64;
    const unsigned char* gwb = W8 + (size_t)(n0 + bn) * K_DIM + bh;

    unsigned char* const sbase = qk_smem;
    const int dA = ss * A_TSTR + sl * A_LSTR;
    const int dB = bn * B_STR + bh;

    const v8f vz = {};
    v8f accH[4][2], accL[4][2];
#pragma unroll
    for (int i = 0; i < 4; ++i)
#pragma unroll
        for (int j = 0; j < 2; ++j) { accH[i][j] = vz; accL[i][j] = vz; }

    qk_stage(ghb, glb, gwb, sbase + AH_OFF + dA, sbase + AL_OFF + dA, sbase + B_OFF + dB);

    for (int kb = 0; kb < NKB; ++kb) {
        unsigned char* cur = sbase + (kb & 1) * BUF_BYTES;
        unsigned char* nxt = sbase + ((kb + 1) & 1) * BUF_BYTES;

        if (kb + 1 < NKB) {
            qk_stage(ghb + (size_t)(kb + 1) * 2048, glb + (size_t)(kb + 1) * 2048,
                     gwb + (size_t)(kb + 1) * 128,
                     nxt + AH_OFF + dA, nxt + AL_OFF + dA, nxt + B_OFF + dB);
#if ASYNC_OK
            asm volatile("s_wait_asynccnt 0xc" ::: "memory");
#endif
        } else {
#if ASYNC_OK
            asm volatile("s_wait_asynccnt 0x0" ::: "memory");
#endif
        }
        __syncthreads();

        v16i bq[2];
#pragma unroll
        for (int j = 0; j < 2; ++j) {
            const unsigned char* bp = cur + B_OFF + (wc * 32 + j * 16 + l16) * B_STR + 16 * lh;
            v4i p0 = *(const v4i*)(bp + 0);
            v4i p1 = *(const v4i*)(bp + 32);
            v4i p2 = *(const v4i*)(bp + 64);
            v4i p3 = *(const v4i*)(bp + 96);
            v8i q01 = __builtin_shufflevector(p0, p1, 0, 1, 2, 3, 4, 5, 6, 7);
            v8i q23 = __builtin_shufflevector(p2, p3, 0, 1, 2, 3, 4, 5, 6, 7);
            bq[j] = __builtin_shufflevector(q01, q23, 0, 1, 2, 3, 4, 5, 6, 7,
                                            8, 9, 10, 11, 12, 13, 14, 15);
        }

#pragma unroll
        for (int i = 0; i < 4; ++i) {
            const int ab = (wr * 4 + i) * A_TSTR + lane * A_LSTR;
            const unsigned char* hp = cur + AH_OFF + ab;
            const unsigned char* lp = cur + AL_OFF + ab;
            v4i h0 = *(const v4i*)(hp + 0);
            v4i h1 = *(const v4i*)(hp + 16);
            v4i h2 = *(const v4i*)(hp + 32);
            v4i h3 = *(const v4i*)(hp + 48);
            v8i hA = __builtin_shufflevector(h0, h1, 0, 1, 2, 3, 4, 5, 6, 7);
            v8i hB = __builtin_shufflevector(h2, h3, 0, 1, 2, 3, 4, 5, 6, 7);
            v16i ah = __builtin_shufflevector(hA, hB, 0, 1, 2, 3, 4, 5, 6, 7,
                                              8, 9, 10, 11, 12, 13, 14, 15);
            v4i l0 = *(const v4i*)(lp + 0);
            v4i l1 = *(const v4i*)(lp + 16);
            v4i l2 = *(const v4i*)(lp + 32);
            v4i l3 = *(const v4i*)(lp + 48);
            v8i lA = __builtin_shufflevector(l0, l1, 0, 1, 2, 3, 4, 5, 6, 7);
            v8i lB = __builtin_shufflevector(l2, l3, 0, 1, 2, 3, 4, 5, 6, 7);
            v16i al = __builtin_shufflevector(lA, lB, 0, 1, 2, 3, 4, 5, 6, 7,
                                              8, 9, 10, 11, 12, 13, 14, 15);
#pragma unroll
            for (int j = 0; j < 2; ++j) {
                accH[i][j] = __builtin_amdgcn_wmma_f32_16x16x128_fp8_fp8(
                    ah, bq[j], (short)0, accH[i][j], false, false);
                accL[i][j] = __builtin_amdgcn_wmma_f32_16x16x128_fp8_fp8(
                    al, bq[j], (short)0, accL[i][j], false, false);
            }
        }
        __syncthreads();
    }

    const float scale = scale_p[0];
#pragma unroll
    for (int i = 0; i < 4; ++i) {
#pragma unroll
        for (int j = 0; j < 2; ++j) {
            const int nn = n0 + wc * 32 + j * 16 + l16;
#pragma unroll
            for (int r = 0; r < 8; ++r) {
                const int mm = blockIdx.y * 128 + wr * 64 + i * 16 + lh * 8 + r;
                Out[(size_t)mm * N_DIM + nn] =
                    (accH[i][j][r] + 0.0625f * accL[i][j][r]) * scale;
            }
        }
    }
}

extern "C" void kernel_launch(void* const* d_in, const int* in_sizes, int n_in,
                              void* d_out, int out_size, void* d_ws, size_t ws_size,
                              hipStream_t stream) {
    const float* x = (const float*)d_in[0];
    const float* W = (const float*)d_in[1];
    float* out = (float*)d_out;

    char* ws = (char*)d_ws;
    float*         scale_p  = (float*)ws;
    float*         partials = (float*)(ws + 256);
    unsigned char* W8  = (unsigned char*)(ws + (1 << 14));
    unsigned char* XHI = (unsigned char*)(ws + (1 << 14) + (16u << 20));
    unsigned char* XLO = (unsigned char*)(ws + (1 << 14) + (48u << 20));

    const int nW = N_DIM * K_DIM;

    qk_abs_partial<<<2048, 256, 0, stream>>>(W, partials, nW);
    qk_abs_final  <<<1,    256, 0, stream>>>(partials, scale_p, 2048, 1.0f / (float)nW);
    qk_quant_w8   <<<4096, 256, 0, stream>>>(W, scale_p, W8, nW / 4);
    qk_split_x8   <<<(M_DIM / 16) * (K_DIM / 128), 256, 0, stream>>>(x, XHI, XLO);

    dim3 grid(N_DIM / 128, M_DIM / 128);
    qk_gemm_fp8<<<grid, 256, 2 * BUF_BYTES, stream>>>(XHI, XLO, W8, scale_p, out);
}
